// GNNClassifier_35605278883841
// MI455X (gfx1250) — matrix operand order repaired and checked by reading, compiled, not run on hardware
//
#include <hip/hip_runtime.h>

typedef __attribute__((ext_vector_type(16))) __bf16 v16bf;
typedef __attribute__((ext_vector_type(8)))  __bf16 v8bf;
typedef __attribute__((ext_vector_type(8)))  float  v8f;

#define N_NODES   100000
#define N_EDGES   3200000
#define F_IN      128
#define HID       256
#define EMB       128
#define N_GRAPHS  2048
#define N_CLASSES 2

// ---------------------------------------------------------------- utilities

__global__ void zero_kernel(float* __restrict__ p, long long n) {
  long long i = (long long)blockIdx.x * blockDim.x + threadIdx.x;
  long long stride = (long long)gridDim.x * blockDim.x;
  for (; i < n; i += stride) p[i] = 0.0f;
}

__global__ void deg_kernel(const int* __restrict__ dst, float* __restrict__ deg, int E) {
  int i = blockIdx.x * blockDim.x + threadIdx.x;
  if (i < E) atomicAdd(&deg[dst[i]], 1.0f);
}

// deg -> dinv = rsqrt(deg + 1)  (self loop included), in place
__global__ void dinv_kernel(float* __restrict__ deg, int n) {
  int i = blockIdx.x * blockDim.x + threadIdx.x;
  if (i < n) deg[i] = rsqrtf(deg[i] + 1.0f);
}

// fp32 -> bf16 copy, 8 elements per thread (b128-vectorized on both sides)
__global__ void cvt_a_kernel(const float* __restrict__ in, __bf16* __restrict__ out,
                             long long n8) {
  long long t = (long long)blockIdx.x * blockDim.x + threadIdx.x;
  if (t >= n8) return;
  const float4 a = ((const float4*)in)[t * 2 + 0];
  const float4 b = ((const float4*)in)[t * 2 + 1];
  v8bf o;
  o[0] = (__bf16)a.x; o[1] = (__bf16)a.y; o[2] = (__bf16)a.z; o[3] = (__bf16)a.w;
  o[4] = (__bf16)b.x; o[5] = (__bf16)b.y; o[6] = (__bf16)b.z; o[7] = (__bf16)b.w;
  ((v8bf*)out)[t] = o;
}

// W [K x M] fp32 -> Wt [M x K] bf16 (transposed). mShift = log2(M).
__global__ void cvt_w_kernel(const float* __restrict__ W, __bf16* __restrict__ Wt,
                             int K, int M, int mShift) {
  int t = blockIdx.x * blockDim.x + threadIdx.x;
  if (t >= K * M) return;
  const int k = t >> mShift;
  const int m = t & (M - 1);
  Wt[(size_t)m * K + k] = (__bf16)W[t];
}

// ------------------------------------------------- WMMA bf16 GEMM  C = A*Wt^T (+bias, relu)
// A:  [N x K] bf16 row major (pre-converted activations)
// Wt: [M x K] bf16 row major (pre-transposed weights)
// C:  [N x M] fp32 row major
// One wave computes a 16x64 output strip (4 WMMA tiles sharing one A fragment).
// N multiple of 16, M multiple of 64, K multiple of 32. grpShift = log2(M/64).
template <bool HAS_BIAS, bool RELU>
__global__ __launch_bounds__(256)
void wmma_gemm_bf16(const __bf16* __restrict__ A, const __bf16* __restrict__ Wt,
                    const float* __restrict__ bias, float* __restrict__ C,
                    int N, int K, int M, int grpShift)
{
  const int wave = blockIdx.x * (blockDim.x >> 5) + (threadIdx.x >> 5);
  const int lane = threadIdx.x & 31;
  const int tileRow = wave >> grpShift;
  const int grp     = wave & ((1 << grpShift) - 1);
  if (tileRow * 16 >= N) return;           // uniform per wave: EXEC stays all-ones

  const int r  = lane & 15;                // A row within tile / B,D column within tile
  const int hi = lane >> 4;                // lane half selector

  const __bf16* arow = A + (size_t)(tileRow * 16 + r) * K;
  const int col0 = (grp << 6) + r;
  const __bf16* brow0 = Wt + (size_t)(col0 +  0) * K;
  const __bf16* brow1 = Wt + (size_t)(col0 + 16) * K;
  const __bf16* brow2 = Wt + (size_t)(col0 + 32) * K;
  const __bf16* brow3 = Wt + (size_t)(col0 + 48) * K;

  v8f acc0 = {}, acc1 = {}, acc2 = {}, acc3 = {};
  for (int k0 = 0; k0 < K; k0 += 32) {
    // A fragment: lane half hi covers K = hi*16 .. hi*16+15 (contiguous, same as B).
    const v8bf alo = *(const v8bf*)(arow + k0 + hi * 16);
    const v8bf ahi = *(const v8bf*)(arow + k0 + hi * 16 + 8);
    const v16bf a = __builtin_shufflevector(alo, ahi,
        0, 1, 2, 3, 4, 5, 6, 7, 8, 9, 10, 11, 12, 13, 14, 15);
    // B fragments: lane half hi covers K = hi*16 .. hi*16+15 of one column,
    // contiguous in the transposed bf16 weight buffer.
    const int kb = k0 + hi * 16;
    const v8bf b0l = *(const v8bf*)(brow0 + kb), b0h = *(const v8bf*)(brow0 + kb + 8);
    const v8bf b1l = *(const v8bf*)(brow1 + kb), b1h = *(const v8bf*)(brow1 + kb + 8);
    const v8bf b2l = *(const v8bf*)(brow2 + kb), b2h = *(const v8bf*)(brow2 + kb + 8);
    const v8bf b3l = *(const v8bf*)(brow3 + kb), b3h = *(const v8bf*)(brow3 + kb + 8);
    const v16bf b0 = __builtin_shufflevector(b0l, b0h, 0,1,2,3,4,5,6,7,8,9,10,11,12,13,14,15);
    const v16bf b1 = __builtin_shufflevector(b1l, b1h, 0,1,2,3,4,5,6,7,8,9,10,11,12,13,14,15);
    const v16bf b2 = __builtin_shufflevector(b2l, b2h, 0,1,2,3,4,5,6,7,8,9,10,11,12,13,14,15);
    const v16bf b3 = __builtin_shufflevector(b3l, b3h, 0,1,2,3,4,5,6,7,8,9,10,11,12,13,14,15);

    acc0 = __builtin_amdgcn_wmma_f32_16x16x32_bf16(false, a, false, b0, (short)0, acc0, false, false);
    acc1 = __builtin_amdgcn_wmma_f32_16x16x32_bf16(false, a, false, b1, (short)0, acc1, false, false);
    acc2 = __builtin_amdgcn_wmma_f32_16x16x32_bf16(false, a, false, b2, (short)0, acc2, false, false);
    acc3 = __builtin_amdgcn_wmma_f32_16x16x32_bf16(false, a, false, b3, (short)0, acc3, false, false);
  }

  // D layout: VGPR i -> row (i + hi*8), col (lane & 15)
  const int ocol = (grp << 6) + (lane & 15);
#pragma unroll
  for (int i = 0; i < 8; ++i) {
    const size_t orow = (size_t)(tileRow * 16 + i + hi * 8);
    float v0 = acc0[i], v1 = acc1[i], v2 = acc2[i], v3 = acc3[i];
    if (HAS_BIAS) {
      v0 += bias[ocol];
      v1 += bias[ocol + 16];
      v2 += bias[ocol + 32];
      v3 += bias[ocol + 48];
    }
    if (RELU) {
      v0 = fmaxf(v0, 0.0f); v1 = fmaxf(v1, 0.0f);
      v2 = fmaxf(v2, 0.0f); v3 = fmaxf(v3, 0.0f);
    }
    float* crow = C + orow * M + ocol;
    crow[0] = v0; crow[16] = v1; crow[32] = v2; crow[48] = v3;
  }
}

// ------------------------------------------------- edge aggregation (scatter-add)
// agg[dst] += h[src] * dinv[src]*dinv[dst]; one thread = 4 features of one edge.
__global__ void agg_kernel(const float* __restrict__ h, const int* __restrict__ src,
                           const int* __restrict__ dst, const float* __restrict__ dinv,
                           float* __restrict__ agg, long long total,
                           int F, int chunkShift, int cmask)
{
  long long tid = (long long)blockIdx.x * blockDim.x + threadIdx.x;
  if (tid >= total) return;
  const int e = (int)(tid >> chunkShift);
  const int f = ((int)tid & cmask) << 2;
  const int s = src[e], d = dst[e];
  const float norm = dinv[s] * dinv[d];
  const float4 hv = *(const float4*)(h + (size_t)s * F + f);
  float* out = agg + (size_t)d * F + f;
  atomicAdd(out + 0, hv.x * norm);
  atomicAdd(out + 1, hv.y * norm);
  atomicAdd(out + 2, hv.z * norm);
  atomicAdd(out + 3, hv.w * norm);
}

// agg = agg + hlin*dinv^2 + b (optional relu), in place. fShift = log2(F).
template <bool RELU>
__global__ void combine_kernel(float* __restrict__ agg, const float* __restrict__ hlin,
                               const float* __restrict__ dinv, const float* __restrict__ bias,
                               long long total, int fShift)
{
  long long tid = (long long)blockIdx.x * blockDim.x + threadIdx.x;
  if (tid >= total) return;
  const int node = (int)(tid >> fShift);
  const int f    = (int)tid & ((1 << fShift) - 1);
  const float di = dinv[node];
  float v = agg[tid] + hlin[tid] * di * di + bias[f];
  if (RELU) v = fmaxf(v, 0.0f);
  agg[tid] = v;
}

// ------------------------------------------------- pooling (EMB = 128: 32 float4 chunks)
__global__ void pool_sum_kernel(const float* __restrict__ h, const int* __restrict__ batch,
                                float* __restrict__ sums, long long total)
{
  long long tid = (long long)blockIdx.x * blockDim.x + threadIdx.x;
  if (tid >= total) return;
  const int node = (int)(tid >> 5);          // EMB/4 = 32 chunks per node
  const int f    = ((int)tid & 31) << 2;
  const int g = batch[node];
  const float4 hv = *(const float4*)(h + (size_t)node * EMB + f);
  float* out = sums + (size_t)g * EMB + f;
  atomicAdd(out + 0, hv.x);
  atomicAdd(out + 1, hv.y);
  atomicAdd(out + 2, hv.z);
  atomicAdd(out + 3, hv.w);
}

__global__ void count_kernel(const int* __restrict__ batch, float* __restrict__ counts, int N) {
  int i = blockIdx.x * blockDim.x + threadIdx.x;
  if (i < N) atomicAdd(&counts[batch[i]], 1.0f);
}

__global__ void pool_div_kernel(const float* __restrict__ sums, const float* __restrict__ counts,
                                float* __restrict__ pooled, int total)
{
  int tid = blockIdx.x * blockDim.x + threadIdx.x;
  if (tid >= total) return;
  pooled[tid] = sums[tid] / fmaxf(counts[tid >> 7], 1.0f);   // EMB = 128
}

// ------------------------------------------------- final tiny classifier layer (C = 2)
__global__ void head_final_kernel(const float* __restrict__ z, const float* __restrict__ W,
                                  const float* __restrict__ b, float* __restrict__ out,
                                  int G, int K)
{
  int idx = blockIdx.x * blockDim.x + threadIdx.x;
  if (idx >= G * N_CLASSES) return;
  const int g = idx >> 1, c = idx & 1;
  float acc = b[c];
  for (int k = 0; k < K; ++k) acc += z[(size_t)g * K + k] * W[(size_t)k * N_CLASSES + c];
  out[idx] = acc;
}

// ---------------------------------------------------------------- driver

static inline int ilog2(int v) { int s = 0; while ((1 << s) < v) ++s; return s; }

extern "C" void kernel_launch(void* const* d_in, const int* in_sizes, int n_in,
                              void* d_out, int out_size, void* d_ws, size_t ws_size,
                              hipStream_t stream) {
  const float* x     = (const float*)d_in[0];
  const int*   ei    = (const int*)d_in[1];
  const int*   batch = (const int*)d_in[2];
  const float* W1 = (const float*)d_in[3];  const float* b1 = (const float*)d_in[4];
  const float* W2 = (const float*)d_in[5];  const float* b2 = (const float*)d_in[6];
  const float* W3 = (const float*)d_in[7];  const float* b3 = (const float*)d_in[8];
  const float* Wc1 = (const float*)d_in[9];  const float* bc1 = (const float*)d_in[10];
  const float* Wc2 = (const float*)d_in[11]; const float* bc2 = (const float*)d_in[12];
  const float* Wc3 = (const float*)d_in[13]; const float* bc3 = (const float*)d_in[14];
  float* out = (float*)d_out;

  const int* src = ei;
  const int* dst = ei + N_EDGES;

  // workspace layout (all offsets stay 16B aligned: every size is a multiple of 16 floats)
  float* ws = (float*)d_ws;
  size_t off = 0;
  float* bufA   = ws + off; off += (size_t)N_NODES * HID;     // h_lin (fp32)
  float* bufB   = ws + off; off += (size_t)N_NODES * HID;     // agg / layer output (fp32)
  float* dinv   = ws + off; off += N_NODES;                   // 100000 (mult of 16)
  float* sums   = ws + off; off += (size_t)N_GRAPHS * EMB;
  float* counts = ws + off; off += N_GRAPHS;
  float* pooled = ws + off; off += (size_t)N_GRAPHS * EMB;
  float* z1     = ws + off; off += (size_t)N_GRAPHS * HID;
  float* z2     = ws + off; off += (size_t)N_GRAPHS * EMB;
  __bf16* Abf   = (__bf16*)(ws + off); off += (size_t)N_NODES * HID / 2;  // bf16 activations
  __bf16* Wt    = (__bf16*)(ws + off); off += (HID * HID) / 2;            // bf16 W^T (<=128KB)

  auto zero = [&](float* p, long long n) {
    zero_kernel<<<4096, 256, 0, stream>>>(p, n);
  };
  auto cvt_a = [&](const float* in, long long n) {         // n multiple of 8
    long long n8 = n >> 3;
    cvt_a_kernel<<<(int)((n8 + 255) / 256), 256, 0, stream>>>(in, Abf, n8);
  };
  auto cvt_w = [&](const float* Wm, int k, int m) {
    cvt_w_kernel<<<(k * m + 255) / 256, 256, 0, stream>>>(Wm, Wt, k, m, ilog2(m));
  };
  auto gemm_plain = [&](float* Cm, int n, int k, int m) {
    int waves = (n / 16) * (m / 64);
    wmma_gemm_bf16<false, false><<<(waves + 7) / 8, 256, 0, stream>>>(
        Abf, Wt, nullptr, Cm, n, k, m, ilog2(m >> 6));
  };
  auto gemm_bias_relu = [&](const float* bias, float* Cm, int n, int k, int m) {
    int waves = (n / 16) * (m / 64);
    wmma_gemm_bf16<true, true><<<(waves + 7) / 8, 256, 0, stream>>>(
        Abf, Wt, bias, Cm, n, k, m, ilog2(m >> 6));
  };
  auto agg = [&](const float* h, float* aggbuf, int F) {
    int chunkShift = ilog2(F >> 2);
    long long total = (long long)N_EDGES << chunkShift;
    agg_kernel<<<(int)((total + 255) / 256), 256, 0, stream>>>(
        h, src, dst, dinv, aggbuf, total, F, chunkShift, (F >> 2) - 1);
  };
  auto combine = [&](float* aggbuf, const float* hlin, const float* bias, int F, bool relu) {
    long long total = (long long)N_NODES * F;
    int blocks = (int)((total + 255) / 256);
    if (relu)
      combine_kernel<true><<<blocks, 256, 0, stream>>>(aggbuf, hlin, dinv, bias, total, ilog2(F));
    else
      combine_kernel<false><<<blocks, 256, 0, stream>>>(aggbuf, hlin, dinv, bias, total, ilog2(F));
  };

  // ---- degrees -> dinv
  zero(dinv, N_NODES);
  deg_kernel<<<(N_EDGES + 255) / 256, 256, 0, stream>>>(dst, dinv, N_EDGES);
  dinv_kernel<<<(N_NODES + 255) / 256, 256, 0, stream>>>(dinv, N_NODES);

  // ---- GCN layer 1: F_IN -> HID, relu
  cvt_w(W1, F_IN, HID);
  cvt_a(x, (long long)N_NODES * F_IN);
  gemm_plain(bufA, N_NODES, F_IN, HID);
  zero(bufB, (long long)N_NODES * HID);
  agg(bufA, bufB, HID);
  combine(bufB, bufA, b1, HID, true);

  // ---- GCN layer 2: HID -> HID, relu
  cvt_w(W2, HID, HID);
  cvt_a(bufB, (long long)N_NODES * HID);
  gemm_plain(bufA, N_NODES, HID, HID);
  zero(bufB, (long long)N_NODES * HID);
  agg(bufA, bufB, HID);
  combine(bufB, bufA, b2, HID, true);

  // ---- GCN layer 3: HID -> EMB, no activation
  cvt_w(W3, HID, EMB);
  cvt_a(bufB, (long long)N_NODES * HID);
  gemm_plain(bufA, N_NODES, HID, EMB);
  zero(bufB, (long long)N_NODES * EMB);
  agg(bufA, bufB, EMB);
  combine(bufB, bufA, b3, EMB, false);

  // ---- mean pooling per graph
  zero(sums, (long long)N_GRAPHS * EMB);
  zero(counts, N_GRAPHS);
  {
    long long total = (long long)N_NODES * (EMB >> 2);
    pool_sum_kernel<<<(int)((total + 255) / 256), 256, 0, stream>>>(bufB, batch, sums, total);
  }
  count_kernel<<<(N_NODES + 255) / 256, 256, 0, stream>>>(batch, counts, N_NODES);
  pool_div_kernel<<<(N_GRAPHS * EMB + 255) / 256, 256, 0, stream>>>(
      sums, counts, pooled, N_GRAPHS * EMB);

  // ---- MLP head
  cvt_w(Wc1, EMB, HID);
  cvt_a(pooled, (long long)N_GRAPHS * EMB);
  gemm_bias_relu(bc1, z1, N_GRAPHS, EMB, HID);                 // 2048x128 @ 128x256, relu
  cvt_w(Wc2, HID, HID / 2);
  cvt_a(z1, (long long)N_GRAPHS * HID);
  gemm_bias_relu(bc2, z2, N_GRAPHS, HID, HID / 2);             // 2048x256 @ 256x128, relu
  head_final_kernel<<<(N_GRAPHS * N_CLASSES + 255) / 256, 256, 0, stream>>>(
      z2, Wc3, bc3, out, N_GRAPHS, HID / 2);                   // 2048x128 @ 128x2
}
